// Seq2Seq_Attention_57440892616831
// MI455X (gfx1250) — matrix operand order repaired and checked by reading, not compiled, not run on hardware
//
#include <hip/hip_runtime.h>
#include <hip/hip_bf16.h>
#include <math.h>

typedef __attribute__((ext_vector_type(16))) __bf16 v16bf;
typedef __attribute__((ext_vector_type(8)))  __bf16 v8bf;
typedef __attribute__((ext_vector_type(8)))  float  v8f;

#define NB   64      // batch
#define NS   50      // src len
#define NT   50      // tgt len
#define NU   1024    // hidden
#define NE   256     // embed
#define NVT  16000   // tgt vocab
#define G3   3072    // 3*U

// ---------------------------------------------------------------------------
// WMMA fragment loaders (wave32, 16x16x32 bf16)
// A (16x32, MxK): lane l holds row m=l&15; element j -> K = j + (j>=8?8:0) + (l>=16?8:0)
// B (32x16, KxN) from transposed storage Bt[N][K]: lane l holds col n=l&15;
//   element j -> K = j + (j>=8?8:0) + (l>=16?8:0)  (same as A)
// C/D (16x16 f32): lane l, vgpr v -> row = v + (l>=16?8:0), col = l&15
// ---------------------------------------------------------------------------
__device__ __forceinline__ v16bf load_fragA(const __bf16* A, int row0, int K, int kc, int lane) {
    int r    = row0 + (lane & 15);
    int koff = (lane >> 4) << 3;              // 0 or 8
    const __bf16* p = A + (size_t)r * K + kc + koff;
    v8bf lo = *(const v8bf*)p;
    v8bf hi = *(const v8bf*)(p + 16);
    v16bf out;
#pragma unroll
    for (int i = 0; i < 8; ++i) { out[i] = lo[i]; out[i + 8] = hi[i]; }
    return out;
}

__device__ __forceinline__ v16bf load_fragB(const __bf16* Bt, int n0, int K, int kc, int lane) {
    int n  = n0 + (lane & 15);
    int kb = kc + ((lane >> 4) << 3);         // 0 or 8
    const __bf16* p = Bt + (size_t)n * K + kb;
    v8bf lo = *(const v8bf*)p;
    v8bf hi = *(const v8bf*)(p + 16);
    v16bf out;
#pragma unroll
    for (int i = 0; i < 8; ++i) { out[i] = lo[i]; out[i + 8] = hi[i]; }
    return out;
}

#define WMMA_BF16(a, b, c) \
    __builtin_amdgcn_wmma_f32_16x16x32_bf16(false, (a), false, (b), (short)0, (c), false, false)

__device__ __forceinline__ float sigmoidf_(float x) { return 1.0f / (1.0f + expf(-x)); }

// ---------------------------------------------------------------------------
// Weight convert + transpose: f32 W[K][N] -> bf16 Wt[N][K]
// ---------------------------------------------------------------------------
__global__ void convt_kernel(const float* __restrict__ W, __bf16* __restrict__ Wt, int K, int N) {
    int idx = blockIdx.x * blockDim.x + threadIdx.x;
    if (idx < K * N) {
        int k = idx / N, n = idx - k * N;
        Wt[(size_t)n * K + k] = (__bf16)W[idx];
    }
}

// init: zero scalar out, seed h buffers from enc_hidden
__global__ void init_kernel(const float* __restrict__ h0, float* __restrict__ hf,
                            __bf16* __restrict__ hb, float* __restrict__ out) {
    int idx = blockIdx.x * blockDim.x + threadIdx.x;
    if (idx < NB * NU) { float v = h0[idx]; hf[idx] = v; hb[idx] = (__bf16)v; }
    if (idx == 0) out[0] = 0.0f;
}

// gather encoder embeddings -> bf16, rows ordered (s*64 + b)
__global__ void gather_enc_kernel(const int* __restrict__ inp, const float* __restrict__ emb,
                                  __bf16* __restrict__ xe) {
    int idx = blockIdx.x * blockDim.x + threadIdx.x;
    if (idx < NB * NS * NE) {
        int e = idx & (NE - 1);
        int r = idx >> 8;                 // r = s*64 + b
        int s = r >> 6, b = r & 63;
        xe[idx] = (__bf16)emb[(size_t)inp[b * NS + s] * NE + e];
    }
}

// ---------------------------------------------------------------------------
// Generic GEMM: C[M][N] = A_bf16[M][K] @ Bt_bf16[N][K]^T + bias
// grid(M/64, N/32), block 128 = 4 waves.  K must be a multiple of 256.
// The 4 waves split M (wave w -> rows bx*64 + 16w..) and SHARE the same two
// n-tiles, so their B loads hit identical addresses (WGP$ dedupes -> L2
// traffic for the streamed weight matrix is minimal: N*K*2 bytes total).
// Outer loop: branchless global_prefetch of the B stream 1KB ahead, then a
// fully unrolled straight-line group of 8 K-chunks (2 WMMAs each).
// ---------------------------------------------------------------------------
__global__ void __launch_bounds__(128)
gemm_bf16_kernel(const __bf16* __restrict__ A, const __bf16* __restrict__ Bt,
                 const float* __restrict__ bias, float* __restrict__ C,
                 int M, int N, int K) {
    int lane = threadIdx.x & 31;
    int wave = threadIdx.x >> 5;
    int m0 = blockIdx.x * 64 + wave * 16;
    int n0 = blockIdx.y * 32;
    int n1 = n0 + 16;
    v8f acc0 = {}, acc1 = {};
    for (int kc0 = 0; kc0 < K; kc0 += 256) {
        // Speculative prefetch of the B stream 512 elements (1KB) ahead.
        // Uniform select (no branch); wraps to 0 near the tail so the
        // address is always in-bounds.
        int kpf = (kc0 + 512 < K) ? (kc0 + 512) : 0;
        __builtin_prefetch(Bt + (size_t)(n0 + (lane & 15)) * K + kpf, 0, 3);
        __builtin_prefetch(Bt + (size_t)(n1 + (lane & 15)) * K + kpf, 0, 3);
#pragma unroll
        for (int kk = 0; kk < 256; kk += 32) {
            int kc = kc0 + kk;
            v16bf a  = load_fragA(A, m0, K, kc, lane);
            v16bf b0 = load_fragB(Bt, n0, K, kc, lane);
            v16bf b1 = load_fragB(Bt, n1, K, kc, lane);
            acc0 = WMMA_BF16(a, b0, acc0);
            acc1 = WMMA_BF16(a, b1, acc1);
        }
    }
    int c0 = n0 + (lane & 15);
    int c1 = n1 + (lane & 15);
    float bv0 = bias ? bias[c0] : 0.0f;
    float bv1 = bias ? bias[c1] : 0.0f;
    int mbase = m0 + ((lane >> 4) << 3);
#pragma unroll
    for (int v = 0; v < 8; ++v) {
        C[(size_t)(mbase + v) * N + c0] = acc0[v] + bv0;
        C[(size_t)(mbase + v) * N + c1] = acc1[v] + bv1;
    }
}

// ---------------------------------------------------------------------------
// Encoder GRU step (fused gi + gh GEMMs + gate math).
// grid(64) = u-tiles, block 128 = 4 waves covering M=64 (B shared per block).
// ---------------------------------------------------------------------------
__global__ void __launch_bounds__(128)
enc_step_kernel(const __bf16* __restrict__ h_bf, const float* __restrict__ h_f,
                const __bf16* __restrict__ xe_t,          // rows t*64.. (64 x 256)
                const __bf16* __restrict__ Wht,           // 3072 x 1024
                const __bf16* __restrict__ Wxt,           // 3072 x 256
                const float*  __restrict__ enc_b,         // 2 x 3072
                float* __restrict__ h_out, __bf16* __restrict__ hbf_out,
                float* __restrict__ enc_out, __bf16* __restrict__ enc_out_bf, int t) {
    int lane = threadIdx.x & 31;
    int wave = threadIdx.x >> 5;
    int u0 = blockIdx.x * 16;
    int m0 = wave * 16;

    v8f az = {}, ar = {}, ahh = {}, ahx = {};
    for (int kc = 0; kc < NU; kc += 32) {                 // gh = h @ Wh
        v16bf a  = load_fragA(h_bf, m0, NU, kc, lane);
        v16bf bz = load_fragB(Wht,        u0, NU, kc, lane);
        v16bf br = load_fragB(Wht, NU   + u0, NU, kc, lane);
        v16bf bh = load_fragB(Wht, 2*NU + u0, NU, kc, lane);
        az  = WMMA_BF16(a, bz, az);
        ar  = WMMA_BF16(a, br, ar);
        ahh = WMMA_BF16(a, bh, ahh);
    }
    for (int kc = 0; kc < NE; kc += 32) {                 // gi = x @ Wx
        v16bf a  = load_fragA(xe_t, m0, NE, kc, lane);
        v16bf bz = load_fragB(Wxt,        u0, NE, kc, lane);
        v16bf br = load_fragB(Wxt, NU   + u0, NE, kc, lane);
        v16bf bh = load_fragB(Wxt, 2*NU + u0, NE, kc, lane);
        az  = WMMA_BF16(a, bz, az);
        ar  = WMMA_BF16(a, br, ar);
        ahx = WMMA_BF16(a, bh, ahx);
    }

    int u = u0 + (lane & 15);
    float b0z = enc_b[u],        b0r = enc_b[NU + u],        b0h = enc_b[2*NU + u];
    float b1z = enc_b[G3 + u],   b1r = enc_b[G3 + NU + u],   b1h = enc_b[G3 + 2*NU + u];
    int mbase = m0 + ((lane >> 4) << 3);
#pragma unroll
    for (int v = 0; v < 8; ++v) {
        int row = mbase + v;
        float z  = sigmoidf_(az[v] + b0z + b1z);
        float r  = sigmoidf_(ar[v] + b0r + b1r);
        float hc = tanhf(ahx[v] + b0h + r * (ahh[v] + b1h));
        float hp = h_f[(size_t)row * NU + u];
        float hn = z * hp + (1.0f - z) * hc;
        h_out[(size_t)row * NU + u]   = hn;
        hbf_out[(size_t)row * NU + u] = (__bf16)hn;
        size_t eo = ((size_t)row * NS + t) * NU + u;
        enc_out[eo]    = hn;
        enc_out_bf[eo] = (__bf16)hn;
    }
}

// ---------------------------------------------------------------------------
// Attention + context + concat(dec embedding) -> x_bf (64 x 1280 bf16)
// grid(64)=batch, block 256.
// ---------------------------------------------------------------------------
__global__ void __launch_bounds__(256)
attn_kernel(const float* __restrict__ q, const float* __restrict__ keys,
            const float* __restrict__ enc_out, const float* __restrict__ Vvec,
            const float* __restrict__ bV, const float* __restrict__ dec_emb,
            const int* __restrict__ targ, __bf16* __restrict__ x_bf, int t) {
    int b = blockIdx.x, tid = threadIdx.x;
    __shared__ float red[256];
    __shared__ float attnw[NS];
    const float* qb = q + (size_t)b * NU;
    const float* kb = keys + (size_t)b * NS * NU;

    for (int s = 0; s < NS; ++s) {
        float p = 0.0f;
        for (int u = tid; u < NU; u += 256)
            p += tanhf(qb[u] + kb[(size_t)s * NU + u]) * Vvec[u];
        red[tid] = p; __syncthreads();
        for (int off = 128; off > 0; off >>= 1) {
            if (tid < off) red[tid] += red[tid + off];
            __syncthreads();
        }
        if (tid == 0) attnw[s] = red[0] + bV[0];
        __syncthreads();
    }
    if (tid == 0) {
        float mx = -1e30f;
        for (int s = 0; s < NS; ++s) mx = fmaxf(mx, attnw[s]);
        float ssum = 0.0f;
        for (int s = 0; s < NS; ++s) { float e = expf(attnw[s] - mx); attnw[s] = e; ssum += e; }
        float inv = 1.0f / ssum;
        for (int s = 0; s < NS; ++s) attnw[s] *= inv;
    }
    __syncthreads();
    const float* eb = enc_out + (size_t)b * NS * NU;
    for (int u = tid; u < NU; u += 256) {
        float c = 0.0f;
        for (int s = 0; s < NS; ++s) c += attnw[s] * eb[(size_t)s * NU + u];
        x_bf[(size_t)b * (NU + NE) + u] = (__bf16)c;
    }
    int tok = targ[b * NT + t];
    for (int e = tid; e < NE; e += 256)
        x_bf[(size_t)b * (NU + NE) + NU + e] = (__bf16)dec_emb[(size_t)tok * NE + e];
}

// ---------------------------------------------------------------------------
// Decoder GRU step (h=0 path: gh = dec_b[1], h_new = (1-z)*hcand) -> dh_bf
// grid(64)=u-tiles, block 128.
// ---------------------------------------------------------------------------
__global__ void __launch_bounds__(128)
dec_step_kernel(const __bf16* __restrict__ x_bf,          // 64 x 1280
                const __bf16* __restrict__ Wxt,           // 3072 x 1280
                const float*  __restrict__ dec_b,         // 2 x 3072
                __bf16* __restrict__ dh_bf) {
    const int K = NU + NE;                                // 1280
    int lane = threadIdx.x & 31;
    int wave = threadIdx.x >> 5;
    int u0 = blockIdx.x * 16;
    int m0 = wave * 16;

    v8f az = {}, ar = {}, ah = {};
    for (int kc = 0; kc < K; kc += 32) {
        v16bf a  = load_fragA(x_bf, m0, K, kc, lane);
        v16bf bz = load_fragB(Wxt,        u0, K, kc, lane);
        v16bf br = load_fragB(Wxt, NU   + u0, K, kc, lane);
        v16bf bh = load_fragB(Wxt, 2*NU + u0, K, kc, lane);
        az = WMMA_BF16(a, bz, az);
        ar = WMMA_BF16(a, br, ar);
        ah = WMMA_BF16(a, bh, ah);
    }
    int u = u0 + (lane & 15);
    float b0z = dec_b[u],      b0r = dec_b[NU + u],      b0h = dec_b[2*NU + u];
    float b1z = dec_b[G3 + u], b1r = dec_b[G3 + NU + u], b1h = dec_b[G3 + 2*NU + u];
    int mbase = m0 + ((lane >> 4) << 3);
#pragma unroll
    for (int v = 0; v < 8; ++v) {
        int row = mbase + v;
        float z  = sigmoidf_(az[v] + b0z + b1z);
        float r  = sigmoidf_(ar[v] + b0r + b1r);
        float hc = tanhf(ah[v] + b0h + r * b1h);
        float hn = (1.0f - z) * hc;
        dh_bf[(size_t)row * NU + u] = (__bf16)hn;
    }
}

// ---------------------------------------------------------------------------
// log_softmax + masked NLL; accumulates mean-over-batch into out[0].
// grid(64)=batch, block 256.
// ---------------------------------------------------------------------------
__global__ void __launch_bounds__(256)
loss_kernel(const float* __restrict__ logits, const int* __restrict__ targ,
            float* __restrict__ out, int t) {
    int b = blockIdx.x, tid = threadIdx.x;
    __shared__ float red[256];
    const float* row = logits + (size_t)b * NVT;
    float mx = -1e30f;
    for (int v = tid; v < NVT; v += 256) mx = fmaxf(mx, row[v]);
    red[tid] = mx; __syncthreads();
    for (int off = 128; off > 0; off >>= 1) {
        if (tid < off) red[tid] = fmaxf(red[tid], red[tid + off]);
        __syncthreads();
    }
    mx = red[0]; __syncthreads();
    float se = 0.0f;
    for (int v = tid; v < NVT; v += 256) se += expf(row[v] - mx);
    red[tid] = se; __syncthreads();
    for (int off = 128; off > 0; off >>= 1) {
        if (tid < off) red[tid] += red[tid + off];
        __syncthreads();
    }
    if (tid == 0) {
        float lse = mx + logf(red[0]);
        int yt = targ[b * NT + t + 1];
        float nll = -(row[yt] - lse);
        float mask = (yt != 0) ? 1.0f : 0.0f;
        atomicAdd(out, nll * mask * (1.0f / (float)NB));
    }
}

// ---------------------------------------------------------------------------
extern "C" void kernel_launch(void* const* d_in, const int* in_sizes, int n_in,
                              void* d_out, int out_size, void* d_ws, size_t ws_size,
                              hipStream_t stream) {
    const int*   inp        = (const int*)  d_in[0];
    const int*   targ       = (const int*)  d_in[1];
    const float* enc_hidden = (const float*)d_in[2];
    const float* enc_emb    = (const float*)d_in[3];
    const float* enc_Wx     = (const float*)d_in[4];
    const float* enc_Wh     = (const float*)d_in[5];
    const float* enc_b      = (const float*)d_in[6];
    const float* W1         = (const float*)d_in[7];
    const float* b1         = (const float*)d_in[8];
    const float* W2         = (const float*)d_in[9];
    const float* b2         = (const float*)d_in[10];
    const float* Vvec       = (const float*)d_in[11];
    const float* bV         = (const float*)d_in[12];
    const float* dec_emb    = (const float*)d_in[13];
    const float* dec_Wx     = (const float*)d_in[14];
    /* d_in[15] = dec_Wh is mathematically dead (decoder GRU gets h==0) */
    const float* dec_b      = (const float*)d_in[16];
    const float* fc_W       = (const float*)d_in[17];
    const float* fc_b       = (const float*)d_in[18];
    float* out = (float*)d_out;

    // ---- workspace carve (all sizes multiples of 256B) ----
    char* w = (char*)d_ws;
    size_t need = 0;
    auto carve = [&](size_t bytes) -> void* {
        void* p = w; size_t a = (bytes + 255) & ~(size_t)255; w += a; need += a; return p;
    };
    __bf16* enc_Wxt   = (__bf16*)carve((size_t)G3 * NE * 2);        // 3072 x 256
    __bf16* enc_Wht   = (__bf16*)carve((size_t)G3 * NU * 2);        // 3072 x 1024
    __bf16* W1t       = (__bf16*)carve((size_t)NU * NU * 2);
    __bf16* W2t       = (__bf16*)carve((size_t)NU * NU * 2);
    __bf16* dec_Wxt   = (__bf16*)carve((size_t)G3 * (NU + NE) * 2); // 3072 x 1280
    __bf16* fc_Wt     = (__bf16*)carve((size_t)NVT * NU * 2);       // 16000 x 1024
    __bf16* xe_bf     = (__bf16*)carve((size_t)NB * NS * NE * 2);
    float*  hfA       = (float*) carve((size_t)NB * NU * 4);
    float*  hfB       = (float*) carve((size_t)NB * NU * 4);
    __bf16* hbA       = (__bf16*)carve((size_t)NB * NU * 2);
    __bf16* hbB       = (__bf16*)carve((size_t)NB * NU * 2);
    float*  enc_out   = (float*) carve((size_t)NB * NS * NU * 4);
    __bf16* enc_outb  = (__bf16*)carve((size_t)NB * NS * NU * 2);
    float*  keys      = (float*) carve((size_t)NB * NS * NU * 4);
    float*  q         = (float*) carve((size_t)NB * NU * 4);
    __bf16* x_bf      = (__bf16*)carve((size_t)NB * (NU + NE) * 2);
    __bf16* dh_bf     = (__bf16*)carve((size_t)NB * NU * 2);
    float*  logits    = (float*) carve((size_t)NB * NVT * 4);
    if (need > ws_size) return;  // workspace too small: bail deterministically

    auto blks = [](size_t n) { return (unsigned)((n + 255) / 256); };

    // ---- weight conversion (bf16, transposed [N][K]) ----
    convt_kernel<<<blks((size_t)NE * G3), 256, 0, stream>>>(enc_Wx, enc_Wxt, NE, G3);
    convt_kernel<<<blks((size_t)NU * G3), 256, 0, stream>>>(enc_Wh, enc_Wht, NU, G3);
    convt_kernel<<<blks((size_t)NU * NU), 256, 0, stream>>>(W1, W1t, NU, NU);
    convt_kernel<<<blks((size_t)NU * NU), 256, 0, stream>>>(W2, W2t, NU, NU);
    convt_kernel<<<blks((size_t)(NU + NE) * G3), 256, 0, stream>>>(dec_Wx, dec_Wxt, NU + NE, G3);
    convt_kernel<<<blks((size_t)NU * NVT), 256, 0, stream>>>(fc_W, fc_Wt, NU, NVT);

    init_kernel<<<blks((size_t)NB * NU), 256, 0, stream>>>(enc_hidden, hfA, hbA, out);
    gather_enc_kernel<<<blks((size_t)NB * NS * NE), 256, 0, stream>>>(inp, enc_emb, xe_bf);

    // ---- encoder scan (ping-pong h buffers) ----
    float*  hf[2] = { hfA, hfB };
    __bf16* hb[2] = { hbA, hbB };
    for (int t = 0; t < NS; ++t) {
        enc_step_kernel<<<NU / 16, 128, 0, stream>>>(
            hb[t & 1], hf[t & 1], xe_bf + (size_t)t * NB * NE,
            enc_Wht, enc_Wxt, enc_b,
            hf[(t + 1) & 1], hb[(t + 1) & 1], enc_out, enc_outb, t);
    }

    // keys = enc_out @ W2 + b2   (M=3200, N=1024, K=1024)
    gemm_bf16_kernel<<<dim3(NB * NS / 64, NU / 32), 128, 0, stream>>>(
        enc_outb, W2t, b2, keys, NB * NS, NU, NU);

    // ---- decoder scan (49 steps) ----
    const __bf16* dh = hb[NS & 1];  // h_last after 50 steps
    for (int t = 0; t < NT - 1; ++t) {
        // q = dh @ W1 + b1
        gemm_bf16_kernel<<<dim3(1, NU / 32), 128, 0, stream>>>(
            dh, W1t, b1, q, NB, NU, NU);
        attn_kernel<<<NB, 256, 0, stream>>>(q, keys, enc_out, Vvec, bV,
                                            dec_emb, targ, x_bf, t);
        dec_step_kernel<<<NU / 16, 128, 0, stream>>>(x_bf, dec_Wxt, dec_b, dh_bf);
        // logits = h_new @ fc_W + fc_b   (M=64, N=16000, K=1024)
        gemm_bf16_kernel<<<dim3(1, NVT / 32), 128, 0, stream>>>(
            dh_bf, fc_Wt, fc_b, logits, NB, NVT, NU);
        loss_kernel<<<NB, 256, 0, stream>>>(logits, targ, out, t);
        dh = dh_bf;
    }
}
